// MultiAtt_44641890075147
// MI455X (gfx1250) — compile-verified, matrix operand order checked by reading, not run on hardware
//
#include <hip/hip_runtime.h>
#include <hip/hip_bf16.h>

typedef __bf16 bf16;
typedef __attribute__((ext_vector_type(16))) __bf16 v16bf;
typedef __attribute__((ext_vector_type(8)))  __bf16 v8bf;
typedef __attribute__((ext_vector_type(4)))  __bf16 v4bf;
typedef __attribute__((ext_vector_type(8)))  float  v8f;
typedef __attribute__((ext_vector_type(4)))  float  v4f;

#define BATCH 8
#define SEQ   1024
#define EMB   768
#define NH    12
#define HD    64
#define QKVD  (3 * EMB)
#define KSTEPS (EMB / 32)
#define SCALE 0.125f   // 64^-0.5, applied POST-softmax per reference

#define WMMA_BF16(a, b, c) \
  __builtin_amdgcn_wmma_f32_16x16x32_bf16(false, (a), false, (b), (short)0, (c), false, false)

__device__ __forceinline__ v8bf ld8(const bf16* p) {
  return *reinterpret_cast<const v8bf*>(p);
}

// 16-byte global -> LDS async copy (CDNA5 GLOBAL_LOAD_ASYNC_TO_LDS_B128,
// tracked by ASYNCcnt). LDS offset = low 32 bits of the generic address.
__device__ __forceinline__ void async_cp16(bf16* lds_dst, const bf16* gsrc) {
  unsigned lo = (unsigned)(unsigned long long)lds_dst;
  asm volatile("global_load_async_to_lds_b128 %0, %1, off"
               :: "v"(lo), "v"(gsrc) : "memory");
}
__device__ __forceinline__ void async_wait0() {
  asm volatile("s_wait_asynccnt 0" ::: "memory");
}

// Build a 16x32 bf16 WMMA operand fragment from a row-major source.
// CDNA5 16-bit operand layout: lanes 0-15 own row (lane&15) with
// K = {0..7, 16..23}; lanes 16-31 own the same rows with K = {8..15, 24..31}.
// Serves A (rows = M) and B (rows = N over pre-transposed data) identically.
__device__ __forceinline__ v16bf load_frag(const bf16* base, int ld, int row0, int k0, int lane) {
  const int r  = lane & 15;
  const int kb = (lane >> 4) << 3;
  const bf16* p = base + (row0 + r) * ld + k0 + kb;
  union { v16bf v; v8bf h[2]; } u;
  u.h[0] = ld8(p);
  u.h[1] = ld8(p + 16);
  return u.v;
}

// ---------------------------------------------------------------- convert
__global__ void k_f32_to_bf16(const float* __restrict__ src, bf16* __restrict__ dst, int n4) {
  int i = blockIdx.x * blockDim.x + threadIdx.x;
  if (i < n4) {
    v4f f = reinterpret_cast<const v4f*>(src)[i];
    v4bf b;
    b[0] = (bf16)f[0]; b[1] = (bf16)f[1]; b[2] = (bf16)f[2]; b[3] = (bf16)f[3];
    reinterpret_cast<v4bf*>(dst)[i] = b;
  }
}

// Shared GEMM k-step: 4 B fragments loaded into distinct registers first so
// the scheduler batches 8 ds_loads + one dscnt drain + 4 back-to-back WMMAs.
__device__ __forceinline__ void gemm_step(const bf16* bp, const v16bf& a, v8f acc[4], int lane) {
  v16bf bfr[4];
#pragma unroll
  for (int et = 0; et < 4; ++et) bfr[et] = load_frag(bp, 32, et * 16, 0, lane);
#pragma unroll
  for (int et = 0; et < 4; ++et) acc[et] = WMMA_BF16(a, bfr[et], acc[et]);
}

// ---------------------------------------------------------------- QKV GEMM
// qkv[t, m] = sum_k x[t,k] * W_qkv[m,k] + b_qkv[m], scattered per head:
//   u==0 -> Q[bh, n, e]   u==1 -> K[bh, n, e]   u==2 -> Vt[bh, e, n]
// Weight panel (64 cols x 32 k) staged in double-buffered LDS via async copies;
// A fragments register double-buffered so WMMAs never wait on global loads.
__global__ __launch_bounds__(256) void k_qkv(const bf16* __restrict__ xb,
                                             const bf16* __restrict__ wb,
                                             const float* __restrict__ bias,
                                             bf16* __restrict__ Qb,
                                             bf16* __restrict__ Kb,
                                             bf16* __restrict__ Vtb) {
  __shared__ bf16 ldsb[2][64 * 32];
  const int tid  = threadIdx.x;
  const int lane = tid & 31;
  const int wave = tid >> 5;
  const int r0 = (blockIdx.y * 8 + wave) * 16;  // token base
  const int c0 = blockIdx.x * 64;               // qkv feature base (64-aligned -> one head)

  // cooperative panel fill: thread -> (col = tid/4, k-octet = (tid%4)*8)
  const int pcol = tid >> 2;
  const int pk   = (tid & 3) << 3;
  const bf16* gp = wb + (c0 + pcol) * EMB + pk;
  bf16* lp[2] = { &ldsb[0][pcol * 32 + pk], &ldsb[1][pcol * 32 + pk] };

  async_cp16(lp[0], gp);
  async_wait0();
  __syncthreads();

  v8f acc[4] = {};
  v16bf a = load_frag(xb, EMB, r0, 0, lane);
  __builtin_prefetch(xb + (r0 + (lane & 15)) * EMB + 64, 0, 3);
#pragma unroll 2
  for (int ks = 0; ks < KSTEPS; ++ks) {
    const int k = ks * 32;
    const int cur = ks & 1;
    const bool more = (ks + 1 < KSTEPS);
    if (more) async_cp16(lp[cur ^ 1], gp + k + 32);    // fill next panel (async)
    v16bf a_n;
    if (more) a_n = load_frag(xb, EMB, r0, k + 32, lane);
    gemm_step(ldsb[cur], a, acc, lane);
    if (more) {
      async_wait0();
      __syncthreads();
      a = a_n;
    }
  }

  const int u    = c0 / EMB;             // 0=Q 1=K 2=V (constant per block)
  const int h    = (c0 % EMB) / HD;      // head (constant per block)
  const int half = lane >> 4;
  const int ncol = lane & 15;
#pragma unroll
  for (int et = 0; et < 4; ++et) {
    const int e  = et * 16 + ncol;       // dim within head
    const float bv = bias[c0 + et * 16 + ncol];
#pragma unroll
    for (int r = 0; r < 8; ++r) {
      const int t = r0 + r + 8 * half;   // C/D layout: lower half rows 0-7, upper 8-15
      const int b = t >> 10, n = t & (SEQ - 1);
      const float val = acc[et][r] + bv;
      const int bh = b * NH + h;
      if (u == 0)      Qb[(bh * SEQ + n) * HD + e]  = (bf16)val;
      else if (u == 1) Kb[(bh * SEQ + n) * HD + e]  = (bf16)val;
      else             Vtb[(bh * HD + e) * SEQ + n] = (bf16)val;   // transposed store
    }
  }
}

// ---------------------------------------------------------------- flash attention
// One wave per (b,h, 16-row q tile). S = Q K^T (no pre-scale), online softmax,
// O = softmax(S) V, epilogue folds SCALE/l. P goes through LDS for C->A relayout.
// All K/V fragments for a 32-key block are loaded up-front so V loads overlap
// the S WMMAs and the softmax VALU work.
__global__ __launch_bounds__(32) void k_attn(const bf16* __restrict__ Qb,
                                             const bf16* __restrict__ Kb,
                                             const bf16* __restrict__ Vtb,
                                             bf16* __restrict__ attb) {
  __shared__ bf16 lds_p[16 * 32];
  const int lane = threadIdx.x & 31;
  const int bh = blockIdx.y;
  const int q0 = blockIdx.x * 16;
  const bf16* Qh = Qb + bh * SEQ * HD;
  const bf16* Kh = Kb + bh * SEQ * HD;
  const bf16* Vh = Vtb + bh * HD * SEQ;   // [e, n]

  const v16bf qa0 = load_frag(Qh, HD, q0, 0,  lane);
  const v16bf qa1 = load_frag(Qh, HD, q0, 32, lane);

  v8f o[4] = {};
  float m[8], l[8];
#pragma unroll
  for (int r = 0; r < 8; ++r) { m[r] = -3.0e38f; l[r] = 0.0f; }
  const int half = lane >> 4;
  const int ncol = lane & 15;

  for (int kb = 0; kb < SEQ; kb += 32) {
    // --- load ALL operands for this 32-key block first (8 x 2 x 16B loads)
    const v16bf kf0 = load_frag(Kh, HD, kb,      0,  lane);
    const v16bf kf1 = load_frag(Kh, HD, kb,      32, lane);
    const v16bf kg0 = load_frag(Kh, HD, kb + 16, 0,  lane);
    const v16bf kg1 = load_frag(Kh, HD, kb + 16, 32, lane);
    v16bf vb[4];
#pragma unroll
    for (int et = 0; et < 4; ++et) vb[et] = load_frag(Vh, SEQ, et * 16, kb, lane);

    // --- S tiles for 32 keys (two 16x16 tiles), K-depth 64 = 2 WMMA each
    v8f s0 = {}, s1 = {};
    s0 = WMMA_BF16(qa0, kf0, s0);
    s0 = WMMA_BF16(qa1, kf1, s0);
    s1 = WMMA_BF16(qa0, kg0, s1);
    s1 = WMMA_BF16(qa1, kg1, s1);

    // --- online softmax update (row stats live inside each 16-lane half)
    float p0[8], p1[8];
#pragma unroll
    for (int r = 0; r < 8; ++r) {
      float t = fmaxf(s0[r], s1[r]);
      t = fmaxf(t, __shfl_xor(t, 1, 32));
      t = fmaxf(t, __shfl_xor(t, 2, 32));
      t = fmaxf(t, __shfl_xor(t, 4, 32));
      t = fmaxf(t, __shfl_xor(t, 8, 32));
      const float nm = fmaxf(m[r], t);
      const float c  = __expf(m[r] - nm);
      const float e0 = __expf(s0[r] - nm);
      const float e1 = __expf(s1[r] - nm);
      float rs = e0 + e1;
      rs += __shfl_xor(rs, 1, 32);
      rs += __shfl_xor(rs, 2, 32);
      rs += __shfl_xor(rs, 4, 32);
      rs += __shfl_xor(rs, 8, 32);
      l[r] = l[r] * c + rs;
      m[r] = nm;
      p0[r] = e0;
      p1[r] = e1;
#pragma unroll
      for (int et = 0; et < 4; ++et) o[et][r] *= c;
    }
    // --- C-layout -> A-layout for P via LDS (16x32 bf16)
    __syncthreads();
#pragma unroll
    for (int r = 0; r < 8; ++r) {
      const int row = r + 8 * half;
      lds_p[row * 32 + ncol]      = (bf16)p0[r];
      lds_p[row * 32 + 16 + ncol] = (bf16)p1[r];
    }
    __syncthreads();
    const v16bf pa = load_frag(lds_p, 32, 0, 0, lane);
    // --- O += P x V : V fragments already resident
#pragma unroll
    for (int et = 0; et < 4; ++et) o[et] = WMMA_BF16(pa, vb[et], o[et]);
  }
  // --- epilogue: att = O * (SCALE / l), merged heads layout [b, n, h*64+e]
  const int bidx = bh / NH;
  const int hcol = (bh % NH) * HD;
#pragma unroll
  for (int et = 0; et < 4; ++et) {
#pragma unroll
    for (int r = 0; r < 8; ++r) {
      const int t = q0 + r + 8 * half;
      const float sc = SCALE / l[r];
      attb[(bidx * SEQ + t) * EMB + hcol + et * 16 + ncol] = (bf16)(o[et][r] * sc);
    }
  }
}

// ---------------------------------------------------------------- output projection
__global__ __launch_bounds__(256) void k_proj(const bf16* __restrict__ attb,
                                              const bf16* __restrict__ wb,
                                              const float* __restrict__ bias,
                                              float* __restrict__ out) {
  __shared__ bf16 ldsb[2][64 * 32];
  const int tid  = threadIdx.x;
  const int lane = tid & 31;
  const int wave = tid >> 5;
  const int r0 = (blockIdx.y * 8 + wave) * 16;
  const int c0 = blockIdx.x * 64;

  const int pcol = tid >> 2;
  const int pk   = (tid & 3) << 3;
  const bf16* gp = wb + (c0 + pcol) * EMB + pk;
  bf16* lp[2] = { &ldsb[0][pcol * 32 + pk], &ldsb[1][pcol * 32 + pk] };

  async_cp16(lp[0], gp);
  async_wait0();
  __syncthreads();

  v8f acc[4] = {};
  v16bf a = load_frag(attb, EMB, r0, 0, lane);
  __builtin_prefetch(attb + (r0 + (lane & 15)) * EMB + 64, 0, 3);
#pragma unroll 2
  for (int ks = 0; ks < KSTEPS; ++ks) {
    const int k = ks * 32;
    const int cur = ks & 1;
    const bool more = (ks + 1 < KSTEPS);
    if (more) async_cp16(lp[cur ^ 1], gp + k + 32);
    v16bf a_n;
    if (more) a_n = load_frag(attb, EMB, r0, k + 32, lane);
    gemm_step(ldsb[cur], a, acc, lane);
    if (more) {
      async_wait0();
      __syncthreads();
      a = a_n;
    }
  }

  const int half = lane >> 4;
  const int ncol = lane & 15;
#pragma unroll
  for (int et = 0; et < 4; ++et) {
    const int mcol = c0 + et * 16 + ncol;
    const float bv = bias[mcol];
#pragma unroll
    for (int r = 0; r < 8; ++r) {
      const int t = r0 + r + 8 * half;
      out[t * EMB + mcol] = acc[et][r] + bv;
    }
  }
}

// ---------------------------------------------------------------- launcher
extern "C" void kernel_launch(void* const* d_in, const int* in_sizes, int n_in,
                              void* d_out, int out_size, void* d_ws, size_t ws_size,
                              hipStream_t stream) {
  (void)in_sizes; (void)n_in; (void)out_size; (void)ws_size;
  const float* x      = (const float*)d_in[0];
  const float* W_qkv  = (const float*)d_in[1];
  const float* b_qkv  = (const float*)d_in[2];
  const float* W_proj = (const float*)d_in[3];
  const float* b_proj = (const float*)d_in[4];
  float* out = (float*)d_out;

  bf16* ws = (bf16*)d_ws;
  size_t o = 0;
  bf16* xb     = ws + o; o += (size_t)BATCH * SEQ * EMB;       // x in bf16
  bf16* wqkvb  = ws + o; o += (size_t)QKVD * EMB;              // W_qkv bf16
  bf16* wprojb = ws + o; o += (size_t)EMB * EMB;               // W_proj bf16
  bf16* Qb     = ws + o; o += (size_t)BATCH * NH * SEQ * HD;   // Q [bh,n,e]
  bf16* Kb     = ws + o; o += (size_t)BATCH * NH * SEQ * HD;   // K [bh,n,e]
  bf16* Vtb    = ws + o; o += (size_t)BATCH * NH * HD * SEQ;   // V^T [bh,e,n]
  bf16* attb   = ws + o; o += (size_t)BATCH * SEQ * EMB;       // merged heads bf16

  const int nx  = BATCH * SEQ * EMB;
  const int nwq = QKVD * EMB;
  const int nwp = EMB * EMB;
  k_f32_to_bf16<<<(nx / 4  + 255) / 256, 256, 0, stream>>>(x,      xb,     nx / 4);
  k_f32_to_bf16<<<(nwq / 4 + 255) / 256, 256, 0, stream>>>(W_qkv,  wqkvb,  nwq / 4);
  k_f32_to_bf16<<<(nwp / 4 + 255) / 256, 256, 0, stream>>>(W_proj, wprojb, nwp / 4);

  // QKV GEMM: 8192 tokens x 2304 features; WG = 128 rows x 64 cols (8 waves)
  k_qkv<<<dim3(QKVD / 64, (BATCH * SEQ) / 128), 256, 0, stream>>>(
      xb, wqkvb, b_qkv, Qb, Kb, Vtb);

  // Flash attention: one wave per (q-tile, b*h)
  k_attn<<<dim3(SEQ / 16, BATCH * NH), 32, 0, stream>>>(Qb, Kb, Vtb, attb);

  // Output projection: 8192 x 768
  k_proj<<<dim3(EMB / 64, (BATCH * SEQ) / 128), 256, 0, stream>>>(
      attb, wprojb, b_proj, out);
}
